// SparseLinear_77738908057636
// MI455X (gfx1250) — compile-verified, matrix operand order checked by reading, not run on hardware
//
#include <hip/hip_runtime.h>
#include <stdint.h>

#define TOKENS 2048
#define IN_F   2048
#define OUT_F  2048
#define NNZ    83968

typedef __attribute__((ext_vector_type(16))) __bf16 v16bf;
typedef __attribute__((ext_vector_type(8)))  float  v8f;
typedef __attribute__((ext_vector_type(4)))  int    v4i;

// Probe-gated CDNA5 async copy path (GLOBAL_LOAD_ASYNC_TO_LDS_B128 + ASYNCcnt)
#if __has_builtin(__builtin_amdgcn_global_load_async_to_lds_b128) && \
    __has_builtin(__builtin_amdgcn_s_wait_asynccnt)
#define USE_ASYNC 1
#else
#define USE_ASYNC 0
#endif

// ---------------------------------------------------------------------------
// Phase 1a: zero the dense fp32 weight buffer in workspace
// ---------------------------------------------------------------------------
__global__ void SL_zero_f32(float* __restrict__ p, int n) {
    int i = blockIdx.x * blockDim.x + threadIdx.x;
    if (i < n) p[i] = 0.0f;
}

// ---------------------------------------------------------------------------
// Phase 1b: scatter COO values into dense W [OUT_F][IN_F] (duplicates add,
// matching segment_sum semantics)
// ---------------------------------------------------------------------------
__global__ void SL_scatter_coo(float* __restrict__ W,
                               const float* __restrict__ vals,
                               const int* __restrict__ rows,
                               const int* __restrict__ cols, int nnz) {
    int k = blockIdx.x * blockDim.x + threadIdx.x;
    if (k < nnz) {
        atomicAdd(&W[(size_t)rows[k] * IN_F + cols[k]], vals[k]);
    }
}

// ---------------------------------------------------------------------------
// Phase 2: split fp32 -> bf16 hi + bf16 lo  (a ~= hi + lo, lo = a - f32(hi))
// ---------------------------------------------------------------------------
__global__ void SL_split_bf16(const float* __restrict__ src,
                              __bf16* __restrict__ hi,
                              __bf16* __restrict__ lo, int n) {
    int i = blockIdx.x * blockDim.x + threadIdx.x;
    if (i < n) {
        float w = src[i];
        __bf16 h = (__bf16)w;                 // RNE f32->bf16
        hi[i] = h;
        lo[i] = (__bf16)(w - (float)h);
    }
}

// ---------------------------------------------------------------------------
// Phase 3: 2048x2048x2048 GEMM out = X * W^T + bias with split-bf16 WMMA.
// Block tile 128(M) x 128(N), K-step 32.  8 waves of 32 (wave32): wave owns
// a 64x32 sub-tile = 4x2 grid of 16x16 f32 accumulators.
//
// Async path: double-buffered LDS, loads issued 2 stages ahead via
// GLOBAL_LOAD_ASYNC_TO_LDS_B128; each stage = 8 async instructions per wave,
// completing in order, so s_wait_asynccnt 8 retires the stage about to be
// consumed while the next stage stays in flight.
// ---------------------------------------------------------------------------
#define TK          32
#define LDS_STRIDE  40   // uint16 per row: 32 data + 8 pad (80 B, 16B-aligned)
#define TILE_U16    (128 * LDS_STRIDE)

__device__ __forceinline__ v16bf SL_load_frag(const uint16_t* __restrict__ tile,
                                              int rowBase, int lane) {
    const int h = lane >> 4;
    const int r = rowBase + (lane & 15);
    const uint16_t* p = tile + r * LDS_STRIDE;
    union { uint4 u[2]; v16bf v; } t;
    t.u[0] = *reinterpret_cast<const uint4*>(p + 8 * h);        // K = [8h, 8h+8)
    t.u[1] = *reinterpret_cast<const uint4*>(p + 16 + 8 * h);   // K = [16+8h, +8)
    return t.v;
}

#if USE_ASYNC
__device__ __forceinline__ void SL_async_cp16(const __bf16* __restrict__ g,
                                              size_t gidx,
                                              uint16_t* l, int lidx) {
    __builtin_amdgcn_global_load_async_to_lds_b128(
        (__attribute__((address_space(1))) v4i*)(g + gidx),
        (__attribute__((address_space(3))) v4i*)(l + lidx),
        0, 0);
}
#endif

__launch_bounds__(256)
__global__ void SL_gemm_split_bf16(const __bf16* __restrict__ Xh,
                                   const __bf16* __restrict__ Xl,
                                   const __bf16* __restrict__ Wh,
                                   const __bf16* __restrict__ Wl,
                                   const float* __restrict__ bias,
                                   float* __restrict__ out) {
#if USE_ASYNC
    __shared__ __align__(16) uint16_t sT[2][4][TILE_U16];   // 80 KB, dbl-buffered
#else
    __shared__ __align__(16) uint16_t sT[1][4][TILE_U16];   // 40 KB
#endif

    const int tid   = threadIdx.x;
    const int lane  = tid & 31;
    const int wv    = tid >> 5;     // 0..7 (wave32)
    const int waveM = wv & 1;       // 2 wave-rows  of 64 tokens
    const int waveN = wv >> 1;      // 4 wave-cols  of 32 features
    const int m0    = blockIdx.y * 128;   // token block
    const int n0    = blockIdx.x * 128;   // out-feature block

    v8f acc[4][2] = {};

#if USE_ASYNC
    // Issue one full stage (4 sub-tiles x 512 16B-chunks) of async copies.
    auto issue_stage = [&](int kt, int buf) {
#pragma unroll
        for (int i = 0; i < 2; ++i) {
            const int chunk = tid + i * 256;      // [0, 512)
            const int r     = chunk >> 2;         // [0, 128)
            const int cc    = chunk & 3;          // 16B column chunk
            const size_t gx = (size_t)(m0 + r) * IN_F + kt + cc * 8;
            const size_t gw = (size_t)(n0 + r) * IN_F + kt + cc * 8;
            const int    ls = r * LDS_STRIDE + cc * 8;
            SL_async_cp16(Xh, gx, sT[buf][0], ls);
            SL_async_cp16(Xl, gx, sT[buf][1], ls);
            SL_async_cp16(Wh, gw, sT[buf][2], ls);
            SL_async_cp16(Wl, gw, sT[buf][3], ls);
        }
    };
    issue_stage(0, 0);
    issue_stage(TK, 1);
#endif

    for (int kt = 0; kt < IN_F; kt += TK) {
#if USE_ASYNC
        const int buf = (kt / TK) & 1;
        if (kt + TK < IN_F) {
            __builtin_amdgcn_s_wait_asynccnt(8);   // current stage retired
        } else {
            __builtin_amdgcn_s_wait_asynccnt(0);   // last stage retired
        }
        __syncthreads();                           // stage visible to all waves
#else
        const int buf = 0;
        __syncthreads();
        // Synchronous staging: 4 tiles x 128 rows x 64B = 32 KB via b128.
#pragma unroll
        for (int i = 0; i < 2; ++i) {
            const int chunk = tid + i * 256;
            const int r     = chunk >> 2;
            const int cc    = chunk & 3;
            const size_t gx = (size_t)(m0 + r) * IN_F + kt + cc * 8;
            const size_t gw = (size_t)(n0 + r) * IN_F + kt + cc * 8;
            const int    ls = r * LDS_STRIDE + cc * 8;
            *reinterpret_cast<uint4*>(&sT[0][0][ls]) =
                *reinterpret_cast<const uint4*>(Xh + gx);
            *reinterpret_cast<uint4*>(&sT[0][1][ls]) =
                *reinterpret_cast<const uint4*>(Xl + gx);
            *reinterpret_cast<uint4*>(&sT[0][2][ls]) =
                *reinterpret_cast<const uint4*>(Wh + gw);
            *reinterpret_cast<uint4*>(&sT[0][3][ls]) =
                *reinterpret_cast<const uint4*>(Wl + gw);
        }
        __syncthreads();
#endif

        v16bf bh[2], bl[2];
#pragma unroll
        for (int nt = 0; nt < 2; ++nt) {
            bh[nt] = SL_load_frag(sT[buf][2], waveN * 32 + nt * 16, lane);
            bl[nt] = SL_load_frag(sT[buf][3], waveN * 32 + nt * 16, lane);
        }
#pragma unroll
        for (int mt = 0; mt < 4; ++mt) {
            const v16bf ah = SL_load_frag(sT[buf][0], waveM * 64 + mt * 16, lane);
            const v16bf al = SL_load_frag(sT[buf][1], waveM * 64 + mt * 16, lane);
#pragma unroll
            for (int nt = 0; nt < 2; ++nt) {
                // (ah+al)*(bh+bl) ~= al*bh + ah*bl + ah*bh  (drop lo*lo)
                acc[mt][nt] = __builtin_amdgcn_wmma_f32_16x16x32_bf16(
                    false, al, false, bh[nt], (short)0, acc[mt][nt], false, false);
                acc[mt][nt] = __builtin_amdgcn_wmma_f32_16x16x32_bf16(
                    false, ah, false, bl[nt], (short)0, acc[mt][nt], false, false);
                acc[mt][nt] = __builtin_amdgcn_wmma_f32_16x16x32_bf16(
                    false, ah, false, bh[nt], (short)0, acc[mt][nt], false, false);
            }
        }

#if USE_ASYNC
        __syncthreads();                          // all waves done reading buf
        if (kt + 2 * TK < IN_F) {
            issue_stage(kt + 2 * TK, buf);        // refill the buffer just freed
        }
#endif
    }

    // Epilogue: C/D 16x16 f32 layout — lane 0-15: N=lane, VGPR v = row v;
    // lane 16-31: N=lane-16, VGPR v = row 8+v.  Fuse bias.
#pragma unroll
    for (int mt = 0; mt < 4; ++mt) {
#pragma unroll
        for (int nt = 0; nt < 2; ++nt) {
            const int row0 = m0 + waveM * 64 + mt * 16 + (lane >> 4) * 8;
            const int col  = n0 + waveN * 32 + nt * 16 + (lane & 15);
            const float bv = bias[col];
#pragma unroll
            for (int v = 0; v < 8; ++v) {
                out[(size_t)(row0 + v) * OUT_F + col] = acc[mt][nt][v] + bv;
            }
        }
    }
}

// ---------------------------------------------------------------------------
// Fallback path (workspace too small): bias init + direct COO atomics
// ---------------------------------------------------------------------------
__global__ void SL_bias_init(float* __restrict__ out,
                             const float* __restrict__ bias, int n) {
    int i = blockIdx.x * blockDim.x + threadIdx.x;
    if (i < n) out[i] = bias[i & (OUT_F - 1)];
}

__global__ void SL_coo_atomic(const float* __restrict__ x,
                              const float* __restrict__ vals,
                              const int* __restrict__ rows,
                              const int* __restrict__ cols,
                              float* __restrict__ out) {
    const int k = blockIdx.x;
    const int r = rows[k];
    const int c = cols[k];
    const float v = vals[k];
    for (int t = threadIdx.x; t < TOKENS; t += blockDim.x) {
        atomicAdd(&out[(size_t)t * OUT_F + r], x[(size_t)t * IN_F + c] * v);
    }
}

// ---------------------------------------------------------------------------
extern "C" void kernel_launch(void* const* d_in, const int* in_sizes, int n_in,
                              void* d_out, int out_size, void* d_ws, size_t ws_size,
                              hipStream_t stream) {
    (void)in_sizes; (void)n_in; (void)out_size;
    const float* x    = (const float*)d_in[0];
    const float* vals = (const float*)d_in[1];
    const float* bias = (const float*)d_in[2];
    const int*   rows = (const int*)d_in[3];
    const int*   cols = (const int*)d_in[4];
    float* out = (float*)d_out;

    const size_t NW   = (size_t)OUT_F * IN_F;   // 4,194,304
    const size_t NX   = (size_t)TOKENS * IN_F;  // 4,194,304
    const size_t need = NW * sizeof(float)         // Wf32
                      + NW * 2 * sizeof(uint16_t)  // Wh, Wl
                      + NX * 2 * sizeof(uint16_t); // Xh, Xl   == 48 MiB

    if (ws_size >= need) {
        float*  Wf = (float*)d_ws;
        __bf16* Wh = (__bf16*)((char*)d_ws + NW * sizeof(float));
        __bf16* Wl = Wh + NW;
        __bf16* Xh = Wl + NW;
        __bf16* Xl = Xh + NX;

        SL_zero_f32   <<<(int)((NW + 255) / 256), 256, 0, stream>>>(Wf, (int)NW);
        SL_scatter_coo<<<(NNZ + 255) / 256,       256, 0, stream>>>(Wf, vals, rows, cols, NNZ);
        SL_split_bf16 <<<(int)((NW + 255) / 256), 256, 0, stream>>>(Wf, Wh, Wl, (int)NW);
        SL_split_bf16 <<<(int)((NX + 255) / 256), 256, 0, stream>>>(x,  Xh, Xl, (int)NX);

        dim3 grid(OUT_F / 128, TOKENS / 128);   // 16 x 16 blocks
        SL_gemm_split_bf16<<<grid, 256, 0, stream>>>(Xh, Xl, Wh, Wl, bias, out);
    } else {
        SL_bias_init <<<(int)((NX + 255) / 256), 256, 0, stream>>>(out, bias, (int)NX);
        SL_coo_atomic<<<NNZ, 256, 0, stream>>>(x, vals, rows, cols, out);
    }
}
